// Block_11974368821691
// MI455X (gfx1250) — compile-verified, matrix operand order checked by reading, not run on hardware
//
#include <hip/hip_runtime.h>

// ---------------------------------------------------------------------------
// Types
// ---------------------------------------------------------------------------
typedef __attribute__((ext_vector_type(8)))  __bf16 v8bf;
typedef __attribute__((ext_vector_type(16))) __bf16 v16bf;
typedef __attribute__((ext_vector_type(8)))  float  v8f;
typedef unsigned short u16;

__device__ __forceinline__ u16 f2bf(float f) {
    unsigned u = __float_as_uint(f);
    u += 0x7fffu + ((u >> 16) & 1u);      // round-to-nearest-even
    return (u16)(u >> 16);
}
__device__ __forceinline__ v8bf ld8(const u16* p) { return *(const v8bf*)p; }
__device__ __forceinline__ v16bf cat8(v8bf lo, v8bf hi) {
    return __builtin_shufflevector(lo, hi, 0,1,2,3,4,5,6,7,8,9,10,11,12,13,14,15);
}
__device__ __forceinline__ v8f zero8() {
    v8f z;
    #pragma unroll
    for (int i = 0; i < 8; i++) z[i] = 0.0f;
    return z;
}
__device__ __forceinline__ float gelu_tanh(float x) {
    float x3 = x * x * x;
    return 0.5f * x * (1.0f + tanhf(0.7978845608028654f * (x + 0.044715f * x3)));
}

// ---------------------------------------------------------------------------
// RMSNorm: out = rmsnorm(x, scale) [+ resid]; f32 out OR bf16 out
// grid = rows, block = 256, D = 2048
// ---------------------------------------------------------------------------
__global__ __launch_bounds__(256) void rmsnorm_kernel(
    const float* __restrict__ x, const float* __restrict__ scale,
    const float* __restrict__ resid, float* __restrict__ outf,
    u16* __restrict__ outb, int D)
{
    const size_t row = blockIdx.x;
    const float* xr = x + row * (size_t)D;
    float ss = 0.0f;
    for (int c = threadIdx.x; c < D; c += 256) { float v = xr[c]; ss += v * v; }
    ss += __shfl_xor(ss, 1);  ss += __shfl_xor(ss, 2);
    ss += __shfl_xor(ss, 4);  ss += __shfl_xor(ss, 8);  ss += __shfl_xor(ss, 16);
    __shared__ float red[8];
    if ((threadIdx.x & 31) == 0) red[threadIdx.x >> 5] = ss;
    __syncthreads();
    float tot = red[0] + red[1] + red[2] + red[3] + red[4] + red[5] + red[6] + red[7];
    float rstd = rsqrtf(tot / (float)D + 1e-6f);
    for (int c = threadIdx.x; c < D; c += 256) {
        float y = xr[c] * rstd * (1.0f + scale[c]);
        if (resid) y += resid[row * (size_t)D + c];
        if (outf) outf[row * (size_t)D + c] = y;
        else      outb[row * (size_t)D + c] = f2bf(y);
    }
}

// ---------------------------------------------------------------------------
// Weight repack: in [nh][D][128] (f32) -> out [(nh*128)][D] (bf16)  (q/k/v)
// ---------------------------------------------------------------------------
__global__ __launch_bounds__(256) void permw_kernel(
    const float* __restrict__ in, u16* __restrict__ out, int D)
{
    size_t i = (size_t)blockIdx.x * 256 + threadIdx.x;   // = ((n*128+h)*D + d)
    int d = (int)(i % D); size_t r = i / D;
    int h = (int)(r % 128); int n = (int)(r / 128);
    out[i] = f2bf(in[((size_t)n * D + d) * 128 + h]);
}

// Transposed convert: out[r*C + c] = in[c*R + r]  (bf16 <- f32)
__global__ __launch_bounds__(256) void transcvt_kernel(
    const float* __restrict__ in, u16* __restrict__ out, int R, int C)
{
    size_t i = (size_t)blockIdx.x * 256 + threadIdx.x;
    int c = (int)(i % C); size_t r = i / C;
    out[i] = f2bf(in[(size_t)c * R + r]);
}

// ---------------------------------------------------------------------------
// Tiled bf16 GEMM:  C[M,N] = A[M,K] * Bt[N,K]^T   (A,Bt bf16 row-major)
// EPI 0: C -> f32.  EPI 2: Cb = bf16( gelu(Gate) * C ).
// block 256 (8 waves), tile 128x128x32, per-wave 64x32 (4x2 WMMA tiles)
// ---------------------------------------------------------------------------
#define LDSTR 40
template<int EPI>
__global__ __launch_bounds__(256) void gemm_bf16_kernel(
    const u16* __restrict__ A, const u16* __restrict__ Bt,
    float* __restrict__ Cf, u16* __restrict__ Cb,
    const float* __restrict__ Gate, int M, int N, int Kd)
{
    __shared__ __align__(16) u16 As[128 * LDSTR];
    __shared__ __align__(16) u16 Bs[128 * LDSTR];
    const int tid  = threadIdx.x;
    const int lane = tid & 31, wave = tid >> 5;
    const int wm = wave >> 2, wn = wave & 3;          // 2 x 4 wave grid
    const int m0 = blockIdx.y * 128, n0 = blockIdx.x * 128;
    const int row = lane & 15, half = lane >> 4;
    const int ar = tid >> 1;                          // staging row 0..127
    const int ac = (tid & 1) * 16;                    // staging col chunk

    v8f acc[4][2];
    #pragma unroll
    for (int i = 0; i < 4; i++) { acc[i][0] = zero8(); acc[i][1] = zero8(); }

    for (int kt = 0; kt < Kd; kt += 32) {
        const u16* ga = A  + (size_t)(m0 + ar) * Kd + kt + ac;
        const u16* gb = Bt + (size_t)(n0 + ar) * Kd + kt + ac;
        __builtin_prefetch(ga + 32, 0, 3);            // global_prefetch next K tile
        __builtin_prefetch(gb + 32, 0, 3);
        *(v8bf*)&As[ar * LDSTR + ac]     = ld8(ga);
        *(v8bf*)&As[ar * LDSTR + ac + 8] = ld8(ga + 8);
        *(v8bf*)&Bs[ar * LDSTR + ac]     = ld8(gb);
        *(v8bf*)&Bs[ar * LDSTR + ac + 8] = ld8(gb + 8);
        __syncthreads();

        v16bf af[4], bfr[2];
        #pragma unroll
        for (int tm = 0; tm < 4; tm++) {
            const u16* p = &As[(wm * 64 + tm * 16 + row) * LDSTR + half * 8];
            af[tm] = cat8(ld8(p), ld8(p + 16));
        }
        #pragma unroll
        for (int tn = 0; tn < 2; tn++) {
            const u16* p = &Bs[(wn * 32 + tn * 16 + row) * LDSTR + half * 8];
            bfr[tn] = cat8(ld8(p), ld8(p + 16));
        }
        #pragma unroll
        for (int tm = 0; tm < 4; tm++)
            #pragma unroll
            for (int tn = 0; tn < 2; tn++)
                acc[tm][tn] = __builtin_amdgcn_wmma_f32_16x16x32_bf16(
                    false, af[tm], false, bfr[tn], (short)0, acc[tm][tn], false, false);
        __syncthreads();
    }

    #pragma unroll
    for (int tm = 0; tm < 4; tm++)
        #pragma unroll
        for (int tn = 0; tn < 2; tn++)
            #pragma unroll
            for (int j = 0; j < 8; j++) {
                int m = m0 + wm * 64 + tm * 16 + j + 8 * half;
                int n = n0 + wn * 32 + tn * 16 + row;
                size_t off = (size_t)m * N + n;
                float v = acc[tm][tn][j];
                if (EPI == 0)       Cf[off] = v;
                else                Cb[off] = f2bf(gelu_tanh(Gate[off]) * v);
            }
}

// ---------------------------------------------------------------------------
// QK-RMSNorm + RoPE (+ query scale):  src [B][T][nh*128] f32
//   -> dst [B][nh][T][128] bf16     grid(nh, T, B), block 128
// ---------------------------------------------------------------------------
__global__ __launch_bounds__(128) void ropeqk_kernel(
    const float* __restrict__ src, const int* __restrict__ pos,
    const float* __restrict__ nscale, u16* __restrict__ dst,
    int T, int nh, float postScale)
{
    const int h = threadIdx.x;
    const int hd = blockIdx.x, t = blockIdx.y, b = blockIdx.z;
    float v = src[(((size_t)(b * T + t)) * nh + hd) * 128 + h];
    float ss = v * v;
    ss += __shfl_xor(ss, 1);  ss += __shfl_xor(ss, 2);
    ss += __shfl_xor(ss, 4);  ss += __shfl_xor(ss, 8);  ss += __shfl_xor(ss, 16);
    __shared__ float red[4];
    __shared__ float sv[128];
    if ((h & 31) == 0) red[h >> 5] = ss;
    __syncthreads();
    float rstd = rsqrtf((red[0] + red[1] + red[2] + red[3]) / 128.0f + 1e-6f);
    float y = v * rstd * (1.0f + nscale[h]);
    sv[h] = y;
    __syncthreads();
    float partner = sv[h ^ 64];
    int idx = h & 63;
    float p = (float)pos[(size_t)b * T + t];
    float ang = p * __powf(10000.0f, -(float)idx / 64.0f);
    float s, c; __sincosf(ang, &s, &c);
    float out = (h < 64) ? (y * c - partner * s) : (y * c + partner * s);
    dst[(((size_t)(b * nh + hd)) * T + t) * 128 + h] = f2bf(out * postScale);
}

// V convert + transpose: vf [B][T][Kh*128] f32 -> vT [B][Kh][128][T] bf16
__global__ __launch_bounds__(256) void vcvt_kernel(
    const float* __restrict__ vf, u16* __restrict__ vT, int T, int Kh)
{
    size_t i = (size_t)blockIdx.x * 256 + threadIdx.x;   // ((b*Kh+kk)*128+h)*T + t
    int t = (int)(i % T); size_t r = i / T;
    int h = (int)(r % 128); r /= 128;
    int kk = (int)(r % Kh); int b = (int)(r / Kh);
    vT[i] = f2bf(vf[((size_t)(b * T + t) * Kh + kk) * 128 + h]);
}

// ---------------------------------------------------------------------------
// Flash attention, soft-cap + causal + sliding-window(512).
// 1 wave / (16 queries, 1 head).  grid(T/16, Nh, B), block 32.
//   qbf [B][Nh][T][128], kbf [B][Kh][T][128], vT [B][Kh][128][T]
//   enc [B][T][Nh*128] bf16
// Soft-cap bounds logits to +/-50 -> fixed-max softmax (exp(l-50)), no
// running max needed.
// ---------------------------------------------------------------------------
__global__ __launch_bounds__(32) void attn_kernel(
    const u16* __restrict__ qbf, const u16* __restrict__ kbf,
    const u16* __restrict__ vT, u16* __restrict__ enc,
    int T, int Nh, int Kh)
{
    const int lane = threadIdx.x;
    const int row = lane & 15, half = lane >> 4;
    const int qt0 = blockIdx.x * 16;
    const int n = blockIdx.y, b = blockIdx.z;
    const int kk = n / (Nh / Kh);

    __shared__ __align__(16) u16 Pl[16 * LDSTR];

    const u16* qbase = qbf + ((size_t)(b * Nh + n) * T + qt0) * 128;
    v16bf aq[4];
    #pragma unroll
    for (int f = 0; f < 4; f++) {
        const u16* p = qbase + row * 128 + f * 32 + half * 8;
        aq[f] = cat8(ld8(p), ld8(p + 16));
    }

    v8f o[8];
    #pragma unroll
    for (int c = 0; c < 8; c++) o[c] = zero8();
    float rowsum[8];
    #pragma unroll
    for (int j = 0; j < 8; j++) rowsum[j] = 0.0f;

    const u16* kbase = kbf + (size_t)(b * Kh + kk) * T * 128;
    const u16* vbase = vT  + (size_t)(b * Kh + kk) * 128 * T;

    int kb_start = qt0 - 511; if (kb_start < 0) kb_start = 0;
    kb_start &= ~31;
    const int kb_last = qt0 + 15;

    for (int kb = kb_start; kb <= kb_last; kb += 32) {
        v8f c0 = zero8(), c1 = zero8();
        #pragma unroll
        for (int f = 0; f < 4; f++) {
            const u16* p0 = kbase + (size_t)(kb + row) * 128 + f * 32 + half * 8;
            v16bf bk0 = cat8(ld8(p0), ld8(p0 + 16));
            c0 = __builtin_amdgcn_wmma_f32_16x16x32_bf16(
                false, aq[f], false, bk0, (short)0, c0, false, false);
            const u16* p1 = kbase + (size_t)(kb + 16 + row) * 128 + f * 32 + half * 8;
            v16bf bk1 = cat8(ld8(p1), ld8(p1 + 16));
            c1 = __builtin_amdgcn_wmma_f32_16x16x32_bf16(
                false, aq[f], false, bk1, (short)0, c1, false, false);
        }
        // soft-cap, mask, exp; stage P (16x32) to LDS for fragment re-layout
        #pragma unroll
        for (int j = 0; j < 8; j++) {
            const int M = j + 8 * half;
            const int qi = qt0 + M;
            int ki = kb + row;
            float l = tanhf(c0[j] * 0.02f) * 50.0f;
            float pr = (ki <= qi && ki > qi - 512) ? __expf(l - 50.0f) : 0.0f;
            rowsum[j] += pr;
            Pl[M * LDSTR + row] = f2bf(pr);
            ki = kb + 16 + row;
            l = tanhf(c1[j] * 0.02f) * 50.0f;
            pr = (ki <= qi && ki > qi - 512) ? __expf(l - 50.0f) : 0.0f;
            rowsum[j] += pr;
            Pl[M * LDSTR + 16 + row] = f2bf(pr);
        }
        __syncthreads();
        const u16* pp = &Pl[row * LDSTR + half * 8];
        v16bf ap = cat8(ld8(pp), ld8(pp + 16));
        #pragma unroll
        for (int c = 0; c < 8; c++) {
            const u16* vp = vbase + (size_t)(c * 16 + row) * T + kb + half * 8;
            v16bf bv = cat8(ld8(vp), ld8(vp + 16));
            o[c] = __builtin_amdgcn_wmma_f32_16x16x32_bf16(
                false, ap, false, bv, (short)0, o[c], false, false);
        }
        __syncthreads();
    }
    #pragma unroll
    for (int j = 0; j < 8; j++) {
        float s = rowsum[j];
        s += __shfl_xor(s, 1); s += __shfl_xor(s, 2);
        s += __shfl_xor(s, 4); s += __shfl_xor(s, 8);
        rowsum[j] = 1.0f / s;
    }
    #pragma unroll
    for (int c = 0; c < 8; c++)
        #pragma unroll
        for (int j = 0; j < 8; j++) {
            int t = qt0 + j + 8 * half;
            size_t off = (size_t)(b * T + t) * (size_t)(Nh * 128) + n * 128 + c * 16 + row;
            enc[off] = f2bf(o[c][j] * rowsum[j]);
        }
}

// ---------------------------------------------------------------------------
// Host-side orchestration
// ---------------------------------------------------------------------------
extern "C" void kernel_launch(void* const* d_in, const int* in_sizes, int n_in,
                              void* d_out, int out_size, void* d_ws, size_t ws_size,
                              hipStream_t stream)
{
    (void)in_sizes; (void)n_in; (void)out_size; (void)ws_size;
    const int Bz = 2, T = 1024, D = 2048, Nh = 16, Kh = 8, HID = 8192;
    const int M = Bz * T;                 // 2048 token rows
    const int NH = Nh * 128, KH = Kh * 128;
    const float SCALAR = 0.08838834764831845f;   // 128^-0.5

    const float* x        = (const float*)d_in[0];
    const int*   segpos   = (const int*)  d_in[1];
    const float* q_kernel = (const float*)d_in[3];
    const float* kv_kern  = (const float*)d_in[4];
    const float* o_kernel = (const float*)d_in[5];
    const float* gate_w   = (const float*)d_in[6];
    const float* up_w     = (const float*)d_in[7];
    const float* down_w   = (const float*)d_in[8];
    const float* pre_a_s  = (const float*)d_in[9];
    const float* post_a_s = (const float*)d_in[10];
    const float* pre_f_s  = (const float*)d_in[11];
    const float* post_f_s = (const float*)d_in[12];
    const float* qns      = (const float*)d_in[13];
    const float* kns      = (const float*)d_in[14];

    char* wp = (char*)d_ws;
    auto carve = [&](size_t bytes) -> char* {
        char* p = wp; wp += (bytes + 255) & ~(size_t)255; return p;
    };
    u16*   hbf     = (u16*)  carve((size_t)M * D * 2);
    u16*   qwT     = (u16*)  carve((size_t)NH * D * 2);
    u16*   kwT     = (u16*)  carve((size_t)KH * D * 2);
    u16*   vwT     = (u16*)  carve((size_t)KH * D * 2);
    u16*   owT     = (u16*)  carve((size_t)D * NH * 2);
    u16*   gwT     = (u16*)  carve((size_t)HID * D * 2);
    u16*   uwT     = (u16*)  carve((size_t)HID * D * 2);
    u16*   dwT     = (u16*)  carve((size_t)D * HID * 2);
    float* qf      = (float*)carve((size_t)M * NH * 4);
    float* kf      = (float*)carve((size_t)M * KH * 4);
    float* vf      = (float*)carve((size_t)M * KH * 4);
    u16*   qb      = (u16*)  carve((size_t)M * NH * 2);
    u16*   kb      = (u16*)  carve((size_t)M * KH * 2);
    u16*   vTb     = (u16*)  carve((size_t)M * KH * 2);
    u16*   encb    = (u16*)  carve((size_t)M * NH * 2);
    float* attnraw = (float*)carve((size_t)M * D * 4);
    float* resid   = (float*)carve((size_t)M * D * 4);
    u16*   h2bf    = (u16*)  carve((size_t)M * D * 2);
    float* gatef   = (float*)carve((size_t)M * HID * 4);
    u16*   actbf   = (u16*)  carve((size_t)M * HID * 2);
    float* mlpraw  = (float*)carve((size_t)M * D * 4);

    // --- pre-attn norm -> bf16 activations ---
    rmsnorm_kernel<<<M, 256, 0, stream>>>(x, pre_a_s, nullptr, nullptr, hbf, D);

    // --- weight repacks to bf16, B-transposed layouts ---
    permw_kernel<<<(int)(((size_t)NH * D) / 256), 256, 0, stream>>>(q_kernel, qwT, D);
    permw_kernel<<<(int)(((size_t)KH * D) / 256), 256, 0, stream>>>(kv_kern, kwT, D);
    permw_kernel<<<(int)(((size_t)KH * D) / 256), 256, 0, stream>>>(
        kv_kern + (size_t)Kh * D * 128, vwT, D);
    transcvt_kernel<<<(int)(((size_t)D * NH) / 256), 256, 0, stream>>>(o_kernel, owT, D, NH);
    transcvt_kernel<<<(int)(((size_t)HID * D) / 256), 256, 0, stream>>>(gate_w, gwT, HID, D);
    transcvt_kernel<<<(int)(((size_t)HID * D) / 256), 256, 0, stream>>>(up_w, uwT, HID, D);
    transcvt_kernel<<<(int)(((size_t)D * HID) / 256), 256, 0, stream>>>(down_w, dwT, D, HID);

    // --- Q / K / V projections (bf16 WMMA GEMMs) ---
    gemm_bf16_kernel<0><<<dim3(NH / 128, M / 128), 256, 0, stream>>>(
        hbf, qwT, qf, nullptr, nullptr, M, NH, D);
    gemm_bf16_kernel<0><<<dim3(KH / 128, M / 128), 256, 0, stream>>>(
        hbf, kwT, kf, nullptr, nullptr, M, KH, D);
    gemm_bf16_kernel<0><<<dim3(KH / 128, M / 128), 256, 0, stream>>>(
        hbf, vwT, vf, nullptr, nullptr, M, KH, D);

    // --- QK-norm + RoPE (+ query scaling); V transpose ---
    ropeqk_kernel<<<dim3(Nh, T, Bz), 128, 0, stream>>>(qf, segpos, qns, qb, T, Nh, SCALAR);
    ropeqk_kernel<<<dim3(Kh, T, Bz), 128, 0, stream>>>(kf, segpos, kns, kb, T, Kh, 1.0f);
    vcvt_kernel<<<(int)(((size_t)M * KH) / 256), 256, 0, stream>>>(vf, vTb, T, Kh);

    // --- attention ---
    attn_kernel<<<dim3(T / 16, Nh, Bz), 32, 0, stream>>>(qb, kb, vTb, encb, T, Nh, Kh);

    // --- output projection + post-attn norm + residual ---
    gemm_bf16_kernel<0><<<dim3(D / 128, M / 128), 256, 0, stream>>>(
        encb, owT, attnraw, nullptr, nullptr, M, D, NH);
    rmsnorm_kernel<<<M, 256, 0, stream>>>(attnraw, post_a_s, x, resid, nullptr, D);

    // --- MLP ---
    rmsnorm_kernel<<<M, 256, 0, stream>>>(resid, pre_f_s, nullptr, nullptr, h2bf, D);
    gemm_bf16_kernel<0><<<dim3(HID / 128, M / 128), 256, 0, stream>>>(
        h2bf, gwT, gatef, nullptr, nullptr, M, HID, D);
    gemm_bf16_kernel<2><<<dim3(HID / 128, M / 128), 256, 0, stream>>>(
        h2bf, uwT, nullptr, actbf, gatef, M, HID, D);
    gemm_bf16_kernel<0><<<dim3(D / 128, M / 128), 256, 0, stream>>>(
        actbf, dwT, mlpraw, nullptr, nullptr, M, D, HID);

    // --- post-ffw norm + residual -> final output (f32) ---
    rmsnorm_kernel<<<M, 256, 0, stream>>>(mlpraw, post_f_s, resid, (float*)d_out, nullptr, D);
}
